// ConditionalARNet_62122406969514
// MI455X (gfx1250) — matrix operand order repaired and checked by reading, not compiled, not run on hardware
//
#include <hip/hip_runtime.h>

typedef __attribute__((ext_vector_type(16))) _Float16 v16h;
typedef __attribute__((ext_vector_type(8)))  _Float16 v8h;
typedef __attribute__((ext_vector_type(8)))  float    v8f;

#define Bsz  4096
#define Nn   29
#define NCc  6
#define CDc  16
#define Dd   22           // NC + CD
#define HDd  64
#define Ll   2
#define NE   (Nn*Nn)      // 841 edges per graph
#define NTIL ((NE+15)/16) // 53 row tiles of 16 edges
#define NW   8            // waves per block (256 threads, wave32)
#define OUTC 24
#define EPSf 1e-5f

// silu/sigmoid with single v_rcp_f32 (no IEEE divide chain)
__device__ __forceinline__ float silu_f(float x){
    return x * __builtin_amdgcn_rcpf(1.f + __expf(-x));
}
__device__ __forceinline__ float sigm_f(float x){
    return __builtin_amdgcn_rcpf(1.f + __expf(-x));
}

extern "C" __global__ __launch_bounds__(256, 1)
void gnn_fused_wmma(const float* __restrict__ x,   const float* __restrict__ ctx,
                    const float* __restrict__ eW1, const float* __restrict__ eb1,
                    const float* __restrict__ eW2, const float* __restrict__ eb2,
                    const float* __restrict__ gW,  const float* __restrict__ gb,
                    const float* __restrict__ lng, const float* __restrict__ lnb,
                    const float* __restrict__ nW1, const float* __restrict__ nb1,
                    const float* __restrict__ nW2, const float* __restrict__ nb2,
                    const float* __restrict__ hW1, const float* __restrict__ hb1,
                    const float* __restrict__ hW2, const float* __restrict__ hb2,
                    float* __restrict__ out)
{
    __shared__ __align__(16) float    zsh [Nn][Dd];      // node features (residual stream)
    __shared__ __align__(16) float    znsh[Nn][Dd];      // layernormed features
    __shared__ __align__(16) float    pish[Nn][HDd];     // source-half projection (+eb1 folded)
    __shared__ __align__(16) float    pjsh[Nn][HDd];     // target-half projection
    __shared__ __align__(16) float    magg[Nn+1][HDd];   // gated aggregate (+1 trash row)
    __shared__ __align__(16) float    hsh [Nn][HDd];     // node-MLP hidden
    __shared__ __align__(16) _Float16 wtsh[HDd][HDd];    // eW2 transposed, f16: wtsh[col][k]
    __shared__ float eb2s[HDd], gws[HDd];
    __shared__ float pooled[Dd], headh[HDd];

    const int b    = blockIdx.x;
    const int tid  = threadIdx.x;
    const int wave = tid >> 5;
    const int lane = tid & 31;

    // ---- build z = concat(x, context) ----
    for (int idx = tid; idx < Nn*Dd; idx += 256) {
        int i = idx / Dd, d = idx % Dd;
        zsh[i][d] = (d < NCc) ? x  [(size_t)b*Nn*NCc + i*NCc + d]
                              : ctx[(size_t)b*Nn*CDc + i*CDc + (d - NCc)];
    }
    __syncthreads();

    for (int l = 0; l < Ll; ++l) {
        const float* eW1l = eW1 + (size_t)l*2*Dd*HDd;
        const float* eW2l = eW2 + (size_t)l*HDd*HDd;
        const float* gWl  = gW  + (size_t)l*HDd;
        const float  gbl  = gb[l];
        const float* eb1l = eb1 + (size_t)l*HDd;
        const float* eb2l = eb2 + (size_t)l*HDd;
        const float* lngl = lng + (size_t)l*Dd;
        const float* lnbl = lnb + (size_t)l*Dd;
        const float* nW1l = nW1 + (size_t)l*(Dd+HDd)*HDd;
        const float* nb1l = nb1 + (size_t)l*HDd;
        const float* nW2l = nW2 + (size_t)l*HDd*Dd;
        const float* nb2l = nb2 + (size_t)l*Dd;

        __builtin_prefetch(eW2l, 0, 3);
        __builtin_prefetch(nW1l, 0, 3);

        // ---- stage biases + f16-transposed eW2 into LDS ----
        if (tid < HDd) { eb2s[tid] = eb2l[tid]; gws[tid] = gWl[tid]; }
        for (int idx = tid; idx < HDd*HDd; idx += 256) {
            int k = idx / HDd, c = idx % HDd;
            wtsh[c][k] = (_Float16)eW2l[idx];
        }
        // ---- pi/pj projections (K=22, tiny -> VALU); fold eb1 into pi; zero aggregate ----
        for (int idx = tid; idx < Nn*HDd; idx += 256) {
            int i = idx / HDd, c = idx % HDd;
            float si = eb1l[c], sj = 0.f;
            for (int d = 0; d < Dd; ++d) {
                float zv = zsh[i][d];
                si += zv * eW1l[d*HDd + c];
                sj += zv * eW1l[(Dd + d)*HDd + c];
            }
            pish[i][c] = si; pjsh[i][c] = sj; magg[i][c] = 0.f;
        }
        // ---- layernorm (one node per thread) ----
        if (tid < Nn) {
            float mu = 0.f;
            for (int d = 0; d < Dd; ++d) mu += zsh[tid][d];
            mu *= (1.f/Dd);
            float var = 0.f;
            for (int d = 0; d < Dd; ++d) { float t = zsh[tid][d]-mu; var += t*t; }
            var *= (1.f/Dd);
            float r = rsqrtf(var + EPSf);
            for (int d = 0; d < Dd; ++d)
                znsh[tid][d] = (zsh[tid][d]-mu)*r*lngl[d] + lnbl[d];
        }
        __syncthreads();

        // ---- preload B fragments (eW2, f16) per wave: kb in {0,1}, nb in {0..3} ----
        // ISA 16-bit B 32x16 layout: lanes 0-15 hold K={0..7,16..23} (col=lane),
        // lanes 16-31 hold K={8..15,24..31} (col=lane-16).
        v16h bf[2][4];
        {
            const int col16 = lane & 15;
            const int koff  = (lane >> 4) ? 8 : 0;
            for (int kb = 0; kb < 2; ++kb)
                for (int nb = 0; nb < 4; ++nb) {
                    const _Float16* base = &wtsh[nb*16 + col16][kb*32 + koff];
                    v8h lo = *(const v8h*)(base);
                    v8h hi = *(const v8h*)(base + 16);
                    v16h f;
                    for (int e = 0; e < 8; ++e) { f[e] = lo[e]; f[8+e] = hi[e]; }
                    bf[kb][nb] = f;
                }
        }

        // ---- edge tiles: m2 = silu(silu(pi_i+pj_j+eb1) @ eW2 + eb2); gated sum over j ----
        for (int t = wave; t < NTIL; t += NW) {
            // A fragments built directly in registers (branchless tail handling).
            // ISA 16-bit A 16x32 layout: lanes 0-15 row=lane, K={0..7,16..23};
            // lanes 16-31 row=lane-16, K={8..15,24..31}.
            const int   row   = lane & 15;
            const int   koffA = (lane >> 4) ? 8 : 0;
            const int   erow  = t*16 + row;
            const bool  valid = (erow < NE);
            const int   ii    = valid ? (erow / Nn) : 0;
            const int   jj    = valid ? (erow % Nn) : 0;

            v16h af[2];
            for (int kb = 0; kb < 2; ++kb) {
                // contiguous 8-float runs -> float4 LDS loads (ds_load_b128)
                const float4* ps = (const float4*)&pish[ii][kb*32 + koffA];
                const float4* qs = (const float4*)&pjsh[jj][kb*32 + koffA];
                float4 p0 = ps[0], p1 = ps[1], p4 = ps[4], p5 = ps[5];
                float4 q0 = qs[0], q1 = qs[1], q4 = qs[4], q5 = qs[5];
                float sv[16];
                sv[0]=p0.x+q0.x; sv[1]=p0.y+q0.y; sv[2]=p0.z+q0.z; sv[3]=p0.w+q0.w;
                sv[4]=p1.x+q1.x; sv[5]=p1.y+q1.y; sv[6]=p1.z+q1.z; sv[7]=p1.w+q1.w;
                sv[8]=p4.x+q4.x; sv[9]=p4.y+q4.y; sv[10]=p4.z+q4.z; sv[11]=p4.w+q4.w;
                sv[12]=p5.x+q5.x; sv[13]=p5.y+q5.y; sv[14]=p5.z+q5.z; sv[15]=p5.w+q5.w;
                v16h f;
                for (int e = 0; e < 16; ++e) {
                    float sx = valid ? sv[e] : 0.f;   // v_cndmask, no branch; silu(0)=0
                    f[e] = (_Float16)silu_f(sx);
                }
                af[kb] = f;
            }

            // GEMM on matrix pipe + fused silu epilogue, keep results in VGPRs
            const int col16 = lane & 15;
            const int mbase = (lane < 16) ? 0 : 8;
            float mval[4][8];
            float gdot[8];
            for (int r = 0; r < 8; ++r) gdot[r] = 0.f;

            for (int nb = 0; nb < 4; ++nb) {
                v8f acc = {};
                acc = __builtin_amdgcn_wmma_f32_16x16x32_f16(false, af[0], false, bf[0][nb],
                                                             (short)0, acc, false, false);
                acc = __builtin_amdgcn_wmma_f32_16x16x32_f16(false, af[1], false, bf[1][nb],
                                                             (short)0, acc, false, false);
                const int col = nb*16 + col16;
                for (int r = 0; r < 8; ++r) {
                    float v = silu_f(acc[r] + eb2s[col]);
                    mval[nb][r] = v;
                    gdot[r] += v * gws[col];
                }
            }
            // butterfly-reduce gate dot across the 16 lanes holding each row
            for (int m = 1; m < 16; m <<= 1)
                for (int r = 0; r < 8; ++r)
                    gdot[r] += __shfl_xor(gdot[r], m);
            // branchless gate + row index (invalid rows -> gate 0, trash row Nn)
            float gate[8];
            int   irow[8];
            for (int r = 0; r < 8; ++r) {
                int e = t*16 + mbase + r;
                float vmask = (e < NE) ? 1.f : 0.f;
                gate[r] = vmask * sigm_f(gdot[r] + gbl);
                irow[r] = (e < NE) ? (e / Nn) : Nn;
            }
            // gated aggregation over neighbors j (full-EXEC ds_add_f32)
            for (int nb = 0; nb < 4; ++nb) {
                const int col = nb*16 + col16;
                for (int r = 0; r < 8; ++r)
                    atomicAdd(&magg[irow[r]][col], gate[r] * mval[nb][r]);
            }
        }
        __syncthreads();

        // ---- node MLP: silu(concat(zn, magg) @ nW1 + nb1) @ nW2 + nb2, residual ----
        for (int idx = tid; idx < Nn*HDd; idx += 256) {
            int i = idx / HDd, c = idx % HDd;
            float s = nb1l[c];
            for (int d = 0; d < Dd;  ++d) s += znsh[i][d] * nW1l[d*HDd + c];
            for (int d = 0; d < HDd; ++d) s += magg[i][d] * nW1l[(Dd + d)*HDd + c];
            hsh[i][c] = silu_f(s);
        }
        __syncthreads();
        for (int idx = tid; idx < Nn*Dd; idx += 256) {
            int i = idx / Dd, d = idx % Dd;
            float s = nb2l[d];
            for (int c = 0; c < HDd; ++c) s += hsh[i][c] * nW2l[c*Dd + d];
            zsh[i][d] += s;
        }
        __syncthreads();
    }

    // ---- mean pool over nodes + head MLP ----
    if (tid < Dd) {
        float s = 0.f;
        for (int i = 0; i < Nn; ++i) s += zsh[i][tid];
        pooled[tid] = s * (1.f/Nn);
    }
    __syncthreads();
    if (tid < HDd) {
        float s = hb1[tid];
        for (int d = 0; d < Dd; ++d) s += pooled[d] * hW1[d*HDd + tid];
        headh[tid] = fmaxf(s, 0.f);
    }
    __syncthreads();

    // ---- output: rows 0..1 = head output reshaped (2,12); rows 2..28 zero-padded ----
    float* ob = out + (size_t)b * Nn * 12;
    for (int idx = tid; idx < (Nn - 2)*12; idx += 256) ob[24 + idx] = 0.f;
    if (tid < OUTC) {
        float s = hb2[tid];
        for (int c = 0; c < HDd; ++c) s += headh[c] * hW2[c*OUTC + tid];
        ob[tid] = s;
    }
}

extern "C" void kernel_launch(void* const* d_in, const int* in_sizes, int n_in,
                              void* d_out, int out_size, void* d_ws, size_t ws_size,
                              hipStream_t stream) {
    (void)in_sizes; (void)n_in; (void)out_size; (void)d_ws; (void)ws_size;
    const float* p[18];
    for (int i = 0; i < 18; ++i) p[i] = (const float*)d_in[i];
    hipLaunchKernelGGL(gnn_fused_wmma, dim3(Bsz), dim3(256), 0, stream,
                       p[0],  p[1],  p[2],  p[3],  p[4],  p[5],
                       p[6],  p[7],  p[8],  p[9],  p[10], p[11],
                       p[12], p[13], p[14], p[15], p[16], p[17],
                       (float*)d_out);
}
